// ASLTransformerModel_62251255988410
// MI455X (gfx1250) — matrix operand order repaired and checked by reading, not compiled, not run on hardware
//
#include <hip/hip_runtime.h>
#include <hip/hip_bf16.h>
#include <math.h>

#define B_ 128
#define S_ 256
#define U_ 512
#define H_ 8
#define D_ 64
#define L_ 4
#define MU_ 2048
#define NC_ 250
#define BS_TOK (B_*S_)

typedef __attribute__((ext_vector_type(16))) __bf16 v16bf;
typedef __attribute__((ext_vector_type(8)))  float  v8f;
typedef int v4i_ __attribute__((vector_size(16)));

union BF16x16 { v16bf v; unsigned int u[8]; };
union F32x8   { v8f   v; float        f[8]; };

// ---- CDNA5 async global->LDS path (guarded; sync b128 fallback) -----------
#if defined(__has_builtin)
#  if __has_builtin(__builtin_amdgcn_global_load_async_to_lds_b128)
#    define USE_ASYNC_LDS 1
#  endif
#endif
#ifndef USE_ASYNC_LDS
#  define USE_ASYNC_LDS 0
#endif
#define AS_G __attribute__((address_space(1)))
#define AS_L __attribute__((address_space(3)))

__device__ __forceinline__ void wait_async_all() {
#if USE_ASYNC_LDS
  asm volatile("s_wait_asynccnt 0x0" ::: "memory");
#endif
}

__device__ __forceinline__ unsigned short f2bf(float x) {
  unsigned int u = __builtin_bit_cast(unsigned int, x);
  unsigned int r = (u + 0x7FFFu + ((u >> 16) & 1u)) >> 16;   // RNE
  return (unsigned short)r;
}
__device__ __forceinline__ unsigned int ldpack(const unsigned short* p) {
  return *(const unsigned int*)p;                            // 2 packed bf16
}
// A-fragment K index map (16-bit A 16x32, ISA 7.12.2)
__device__ __forceinline__ int amap(int j, int half) {
  return (j < 4) ? (8*half + 2*j) : (16 + 8*half + 2*(j-4));
}
// B-fragment K index map (16-bit B 32x16), same K order as amap
__device__ __forceinline__ int bmap(int j, int half) { return (j < 4) ? (8*half + 2*j) : (16 + 8*half + 2*(j-4)); }

__device__ __forceinline__ float gelu_exact(float v) {
  return 0.5f * v * (1.0f + erff(v * 0.70710678118654752440f));
}

// ---------------------------------------------------------------------------
// bf16 full-tile GEMM (M%128==0, N%128==0, K%32==0 by construction).
// A: bf16 [M][K] row-major; Bt: bf16 [N][K] (pre-transposed weights).
// Double-buffered LDS staging; async global->LDS when available.
// mode 0: C = [resid +] act(A@B + bias*bias_scale)
// mode 2: landmark combine: C (+)= softmax(lmw)[group] * (miss ? empty : v)
// ---------------------------------------------------------------------------
#define LDS_STRIDE 40   // 32 bf16 + 8 pad: 16B-aligned rows, conflict-free

__device__ __forceinline__ void stage_tile_128x32(
    const unsigned short* __restrict__ g, long ldg, unsigned short* l)
{
  const int tid = threadIdx.x;
  #pragma unroll
  for (int it = 0; it < 2; ++it) {
    int c = it * 256 + tid;            // 512 x 16B chunks
    int row = c >> 2, part = (c & 3) * 8;
    const unsigned short* gp = g + (long)row * ldg + part;
    unsigned short* lp = l + row * LDS_STRIDE + part;
#if USE_ASYNC_LDS
    __builtin_amdgcn_global_load_async_to_lds_b128((AS_G v4i_*)gp, (AS_L v4i_*)lp, 0, 0);
#else
    *(uint4*)lp = *(const uint4*)gp;
#endif
  }
}

__global__ __launch_bounds__(256) void gemm_bf16_kernel(
    const unsigned short* __restrict__ A, int lda,
    const unsigned short* __restrict__ Bt, int ldb,
    float* __restrict__ C, unsigned short* __restrict__ Cbf, int ldc,
    int M, int N, int K,
    const float* __restrict__ bias, float bias_scale,
    const float* __restrict__ resid,
    int act_gelu, int mode, int group,
    const float* __restrict__ lmw,
    const float* __restrict__ emptyv,
    const float* __restrict__ missf)
{
  __shared__ __align__(16) unsigned short As[2][128 * LDS_STRIDE];
  __shared__ __align__(16) unsigned short Bs[2][128 * LDS_STRIDE];
  const int tid  = threadIdx.x;
  const int lane = tid & 31;
  const int wv   = tid >> 5;
  const int half = lane >> 4;
  const int lm   = lane & 15;
  const int m0   = blockIdx.y * 128;
  const int n0   = blockIdx.x * 128;
  const int wm   = (wv & 3) * 32;   // 4 waves along M
  const int wn   = (wv >> 2) * 64;  // 2 waves along N

  const unsigned short* Abase = A  + (long)m0 * lda;
  const unsigned short* Bbase = Bt + (long)n0 * ldb;

  F32x8 acc[2][4];
  #pragma unroll
  for (int mt = 0; mt < 2; ++mt)
    #pragma unroll
    for (int nt = 0; nt < 4; ++nt)
      #pragma unroll
      for (int r = 0; r < 8; ++r) acc[mt][nt].f[r] = 0.0f;

  const int nk = K >> 5;
  stage_tile_128x32(Abase, lda, As[0]);
  stage_tile_128x32(Bbase, ldb, Bs[0]);
  wait_async_all();
  __syncthreads();

  for (int kt = 0; kt < nk; ++kt) {
    const int cur = kt & 1;
    if (kt + 1 < nk) {                 // prefetch next k-tile into other buffer
      stage_tile_128x32(Abase + (kt + 1) * 32, lda, As[1 - cur]);
      stage_tile_128x32(Bbase + (kt + 1) * 32, ldb, Bs[1 - cur]);
    }
    BF16x16 afrag[2], bfrag[4];
    #pragma unroll
    for (int mt = 0; mt < 2; ++mt) {
      int row = wm + mt * 16 + lm;
      #pragma unroll
      for (int j = 0; j < 8; ++j)
        afrag[mt].u[j] = ldpack(&As[cur][row * LDS_STRIDE + amap(j, half)]);
    }
    #pragma unroll
    for (int nt = 0; nt < 4; ++nt) {
      int col = wn + nt * 16 + lm;
      #pragma unroll
      for (int j = 0; j < 8; ++j)
        bfrag[nt].u[j] = ldpack(&Bs[cur][col * LDS_STRIDE + bmap(j, half)]);
    }
    #pragma unroll
    for (int mt = 0; mt < 2; ++mt)
      #pragma unroll
      for (int nt = 0; nt < 4; ++nt)
        acc[mt][nt].v = __builtin_amdgcn_wmma_f32_16x16x32_bf16(
            false, afrag[mt].v, false, bfrag[nt].v, (short)0, acc[mt][nt].v, false, false);
    if (kt + 1 < nk) {
      wait_async_all();
      __syncthreads();
    }
  }

  float wgt = 1.0f;
  if (mode == 2) {
    float w0 = lmw[0], w1 = lmw[1], w2 = lmw[2];
    float mx = fmaxf(w0, fmaxf(w1, w2));
    float e0 = __expf(w0 - mx), e1 = __expf(w1 - mx), e2 = __expf(w2 - mx);
    float s = e0 + e1 + e2;
    wgt = ((group == 0) ? e0 : (group == 1) ? e1 : e2) / s;
  }
  #pragma unroll
  for (int mt = 0; mt < 2; ++mt)
    #pragma unroll
    for (int nt = 0; nt < 4; ++nt)
      #pragma unroll
      for (int r = 0; r < 8; ++r) {
        int row = m0 + wm + mt * 16 + r + 8 * half;   // C-layout m = r + 8*half
        int col = n0 + wn + nt * 16 + lm;             //          n = lane&15
        float v = acc[mt][nt].f[r];
        if (bias) v += bias[col] * bias_scale;
        if (act_gelu) v = gelu_exact(v);
        if (mode == 2) {
          float sel  = (missf[row * 3 + group] != 0.0f) ? emptyv[col] : v;
          float prev = (group == 0) ? 0.0f : C[(long)row * ldc + col];
          v = prev + wgt * sel;
        } else if (resid) {
          v += resid[(long)row * ldc + col];
        }
        if (C)   C[(long)row * ldc + col]   = v;
        if (Cbf) Cbf[(long)row * ldc + col] = f2bf(v);
      }
}

// ---------------------------------------------------------------------------
// Edge GEMM (ragged K / N): f32 A and B, clamped (branch-free) tile loads.
// Used only for landmark W1 projections (K=80/42/10) and the classifier.
// ---------------------------------------------------------------------------
#define LDT 34

__global__ __launch_bounds__(256) void gemm_edge_kernel(
    const float* __restrict__ A, int lda,
    const float* __restrict__ Bw, int ldb,
    float* __restrict__ C, unsigned short* __restrict__ Cbf, int ldc,
    int M, int N, int K,
    const float* __restrict__ bias, int act_gelu)
{
  __shared__ __align__(16) unsigned short As[128 * LDT];
  __shared__ __align__(16) unsigned short Bs[128 * LDT];
  const int tid  = threadIdx.x;
  const int lane = tid & 31;
  const int wv   = tid >> 5;
  const int half = lane >> 4;
  const int lm   = lane & 15;
  const int m0   = blockIdx.y * 128;
  const int n0   = blockIdx.x * 128;
  const int wm   = (wv & 3) * 32;
  const int wn   = (wv >> 2) * 64;

  F32x8 acc[2][4];
  #pragma unroll
  for (int mt = 0; mt < 2; ++mt)
    #pragma unroll
    for (int nt = 0; nt < 4; ++nt)
      #pragma unroll
      for (int r = 0; r < 8; ++r) acc[mt][nt].f[r] = 0.0f;

  for (int k0 = 0; k0 < K; k0 += 32) {
    __syncthreads();
    #pragma unroll
    for (int it = 0; it < 16; ++it) {         // A tile 128x32 (clamped loads)
      int idx = it * 256 + tid;
      int r = idx >> 5, c = idx & 31;
      int gr = m0 + r, gc = k0 + c;
      int grl = (gr < M) ? gr : (M - 1);
      int gcl = (gc < K) ? gc : (K - 1);
      float v = A[(long)grl * lda + gcl];
      if (gr >= M || gc >= K) v = 0.0f;
      As[r * LDT + c] = f2bf(v);
    }
    #pragma unroll
    for (int it = 0; it < 16; ++it) {         // B tile 32x128 -> [n][k]
      int idx = it * 256 + tid;
      int k = idx >> 7, n = idx & 127;
      int gk = k0 + k, gn = n0 + n;
      int gkl = (gk < K) ? gk : (K - 1);
      int gnl = (gn < N) ? gn : (N - 1);
      float v = Bw[(long)gkl * ldb + gnl];
      if (gk >= K || gn >= N) v = 0.0f;
      Bs[n * LDT + k] = f2bf(v);
    }
    __syncthreads();

    BF16x16 afrag[2], bfrag[4];
    #pragma unroll
    for (int mt = 0; mt < 2; ++mt) {
      int row = wm + mt * 16 + lm;
      #pragma unroll
      for (int j = 0; j < 8; ++j) afrag[mt].u[j] = ldpack(&As[row * LDT + amap(j, half)]);
    }
    #pragma unroll
    for (int nt = 0; nt < 4; ++nt) {
      int col = wn + nt * 16 + lm;
      #pragma unroll
      for (int j = 0; j < 8; ++j) bfrag[nt].u[j] = ldpack(&Bs[col * LDT + bmap(j, half)]);
    }
    #pragma unroll
    for (int mt = 0; mt < 2; ++mt)
      #pragma unroll
      for (int nt = 0; nt < 4; ++nt)
        acc[mt][nt].v = __builtin_amdgcn_wmma_f32_16x16x32_bf16(
            false, afrag[mt].v, false, bfrag[nt].v, (short)0, acc[mt][nt].v, false, false);
  }

  #pragma unroll
  for (int mt = 0; mt < 2; ++mt)
    #pragma unroll
    for (int nt = 0; nt < 4; ++nt)
      #pragma unroll
      for (int r = 0; r < 8; ++r) {
        int row = m0 + wm + mt * 16 + r + 8 * half;
        int col = n0 + wn + nt * 16 + lm;
        if (row < M && col < N) {
          float v = acc[mt][nt].f[r];
          if (bias) v += bias[col];
          if (act_gelu) v = gelu_exact(v);
          if (C)   C[(long)row * ldc + col]   = v;
          if (Cbf) Cbf[(long)row * ldc + col] = f2bf(v);
        }
      }
}

// ---------------------------------------------------------------------------
// Flash-style attention per (b,h). bf16 q/k/v in, bf16 z out. Softmax scale
// is folded into Wq/bq. 8 waves x (2 sequential 16-row WMMA blocks); K/V
// staged in LDS in 128-key halves; online softmax; P relaid via per-wave LDS.
// ---------------------------------------------------------------------------
__global__ __launch_bounds__(256) void attn_kernel(
    const unsigned short* __restrict__ Q, const unsigned short* __restrict__ Kf,
    const unsigned short* __restrict__ Vf, const float* __restrict__ nef,
    unsigned short* __restrict__ Z)
{
  __shared__ __align__(16) unsigned short Ks[128 * 66];   // [key][d]
  __shared__ __align__(16) unsigned short Vt[64 * 130];   // [d][key]
  __shared__ float mb[S_];
  __shared__ __align__(16) unsigned short Pscr[8][16 * 34];

  const int b    = blockIdx.x / H_;
  const int h    = blockIdx.x % H_;
  const int tid  = threadIdx.x;
  const int lane = tid & 31;
  const int wv   = tid >> 5;
  const int half = lane >> 4;
  const int lm   = lane & 15;

  mb[tid] = (nef[b * S_ + tid] != -1.0f) ? 0.0f : -1.0e9f;

  for (int rb = 0; rb < 2; ++rb) {
    const int row_base = wv * 32 + rb * 16;
    BF16x16 qf[2];
    {
      int row = row_base + lm;
      long qoff = ((long)(b * S_ + row)) * U_ + h * D_;
      #pragma unroll
      for (int c = 0; c < 2; ++c)
        #pragma unroll
        for (int j = 0; j < 8; ++j)
          qf[c].u[j] = ldpack(Q + qoff + c * 32 + amap(j, half));
    }
    float rmax[8], rsum[8];
    #pragma unroll
    for (int r = 0; r < 8; ++r) { rmax[r] = -3.0e38f; rsum[r] = 0.0f; }
    F32x8 zacc[4];
    #pragma unroll
    for (int dt = 0; dt < 4; ++dt)
      #pragma unroll
      for (int r = 0; r < 8; ++r) zacc[dt].f[r] = 0.0f;

    for (int kh = 0; kh < 2; ++kh) {
      __syncthreads();
      #pragma unroll
      for (int it = 0; it < 16; ++it) {   // stage 128 keys x 64 d (u32 pairs)
        int idx = it * 256 + tid;
        int s = idx >> 5;
        int dp = (idx & 31) * 2;
        int sg = kh * 128 + s;
        long gb = ((long)(b * S_ + sg)) * U_ + h * D_ + dp;
        unsigned int kv = ldpack(Kf + gb);
        *(unsigned int*)&Ks[s * 66 + dp] = kv;
        unsigned int vv = ldpack(Vf + gb);
        Vt[dp * 130 + s]       = (unsigned short)(vv & 0xFFFFu);
        Vt[(dp + 1) * 130 + s] = (unsigned short)(vv >> 16);
      }
      __syncthreads();

      for (int kc = 0; kc < 4; ++kc) {
        const int key0  = kc * 32;
        const int key0g = kh * 128 + key0;
        F32x8 sc[2];
        #pragma unroll
        for (int nt = 0; nt < 2; ++nt)
          #pragma unroll
          for (int r = 0; r < 8; ++r) sc[nt].f[r] = 0.0f;
        #pragma unroll
        for (int c = 0; c < 2; ++c)
          #pragma unroll
          for (int nt = 0; nt < 2; ++nt) {
            BF16x16 bk;
            int key = key0 + nt * 16 + lm;
            #pragma unroll
            for (int j = 0; j < 8; ++j)
              bk.u[j] = ldpack(&Ks[key * 66 + c * 32 + bmap(j, half)]);
            sc[nt].v = __builtin_amdgcn_wmma_f32_16x16x32_bf16(
                false, qf[c].v, false, bk.v, (short)0, sc[nt].v, false, false);
          }
        #pragma unroll
        for (int nt = 0; nt < 2; ++nt) {
          float bn = mb[key0g + nt * 16 + lm];
          #pragma unroll
          for (int r = 0; r < 8; ++r) sc[nt].f[r] += bn;
        }
        float cm[8];
        #pragma unroll
        for (int r = 0; r < 8; ++r) cm[r] = fmaxf(sc[0].f[r], sc[1].f[r]);
        #pragma unroll
        for (int off = 1; off < 16; off <<= 1)
          #pragma unroll
          for (int r = 0; r < 8; ++r) cm[r] = fmaxf(cm[r], __shfl_xor(cm[r], off, 32));
        float corr[8];
        #pragma unroll
        for (int r = 0; r < 8; ++r) {
          float nm = fmaxf(rmax[r], cm[r]);
          corr[r] = __expf(rmax[r] - nm);
          rmax[r] = nm;
        }
        float cs[8];
        #pragma unroll
        for (int r = 0; r < 8; ++r) cs[r] = 0.0f;
        #pragma unroll
        for (int nt = 0; nt < 2; ++nt)
          #pragma unroll
          for (int r = 0; r < 8; ++r) {
            float p = __expf(sc[nt].f[r] - rmax[r]);
            sc[nt].f[r] = p;
            cs[r] += p;
          }
        #pragma unroll
        for (int off = 1; off < 16; off <<= 1)
          #pragma unroll
          for (int r = 0; r < 8; ++r) cs[r] += __shfl_xor(cs[r], off, 32);
        #pragma unroll
        for (int r = 0; r < 8; ++r) rsum[r] = rsum[r] * corr[r] + cs[r];
        #pragma unroll
        for (int dt = 0; dt < 4; ++dt)
          #pragma unroll
          for (int r = 0; r < 8; ++r) zacc[dt].f[r] *= corr[r];
        unsigned short* ps = &Pscr[wv][0];
        #pragma unroll
        for (int nt = 0; nt < 2; ++nt)
          #pragma unroll
          for (int r = 0; r < 8; ++r)
            ps[(r + 8 * half) * 34 + nt * 16 + lm] = f2bf(sc[nt].f[r]);
        BF16x16 pf;
        #pragma unroll
        for (int j = 0; j < 8; ++j) pf.u[j] = ldpack(&ps[lm * 34 + amap(j, half)]);
        #pragma unroll
        for (int dt = 0; dt < 4; ++dt) {
          BF16x16 bv;
          int d = dt * 16 + lm;
          #pragma unroll
          for (int j = 0; j < 8; ++j)
            bv.u[j] = ldpack(&Vt[d * 130 + key0 + bmap(j, half)]);
          zacc[dt].v = __builtin_amdgcn_wmma_f32_16x16x32_bf16(
              false, pf.v, false, bv.v, (short)0, zacc[dt].v, false, false);
        }
      }
    }
    #pragma unroll
    for (int dt = 0; dt < 4; ++dt)
      #pragma unroll
      for (int r = 0; r < 8; ++r) {
        int row = row_base + r + 8 * half;
        int d   = dt * 16 + lm;
        Z[((long)(b * S_ + row)) * U_ + h * D_ + d] = f2bf(zacc[dt].f[r] / rsum[r]);
      }
  }
}

// ---------------------------------------------------------------------------
// Helper kernels
// ---------------------------------------------------------------------------
__global__ void preprocess_kernel(
    const float* __restrict__ frames,
    const float* __restrict__ lmean, const float* __restrict__ lstd,
    const float* __restrict__ hmean, const float* __restrict__ hstd,
    const float* __restrict__ pmean, const float* __restrict__ pstd,
    float* __restrict__ fl, float* __restrict__ fh, float* __restrict__ fp,
    float* __restrict__ miss)
{
  int bs = blockIdx.x * blockDim.x + threadIdx.x;
  if (bs >= BS_TOK) return;
  const float* fr = frames + (long)bs * 66 * 3;
  float sum = 0.0f;
  for (int j = 0; j < 40; ++j)
    for (int c = 0; c < 2; ++c) {
      float t = fr[j * 3 + c];
      float v = (t == 0.0f) ? 0.0f : (t - lmean[j * 2 + c]) / lstd[j * 2 + c];
      fl[(long)bs * 80 + j * 2 + c] = v; sum += v;
    }
  miss[bs * 3 + 0] = (sum == 0.0f) ? 1.0f : 0.0f;
  sum = 0.0f;
  for (int j = 0; j < 21; ++j)
    for (int c = 0; c < 2; ++c) {
      float t = fr[(40 + j) * 3 + c];
      float v = (t == 0.0f) ? 0.0f : (t - hmean[j * 2 + c]) / hstd[j * 2 + c];
      fh[(long)bs * 42 + j * 2 + c] = v; sum += v;
    }
  miss[bs * 3 + 1] = (sum == 0.0f) ? 1.0f : 0.0f;
  sum = 0.0f;
  for (int j = 0; j < 5; ++j)
    for (int c = 0; c < 2; ++c) {
      float t = fr[(61 + j) * 3 + c];
      float v = (t == 0.0f) ? 0.0f : (t - pmean[j * 2 + c]) / pstd[j * 2 + c];
      fp[(long)bs * 10 + j * 2 + c] = v; sum += v;
    }
  miss[bs * 3 + 2] = (sum == 0.0f) ? 1.0f : 0.0f;
}

// W [K][N] f32 -> Wt [N][K] bf16 (optionally scaled)
__global__ void wtconv_kernel(const float* __restrict__ W, unsigned short* __restrict__ Wt,
                              int K, int N, float scale) {
  int idx = blockIdx.x * 256 + threadIdx.x;
  if (idx >= K * N) return;
  int n = idx / K, k = idx - n * K;
  Wt[(long)n * K + k] = f2bf(W[(long)k * N + n] * scale);
}

// W [H][U][D] f32 -> Wt [H*D][U] bf16 (packed-transposed QKV weight)
__global__ void wqkvconv_kernel(const float* __restrict__ W, unsigned short* __restrict__ Wt,
                                float scale) {
  int idx = blockIdx.x * 256 + threadIdx.x;
  if (idx >= U_ * U_) return;
  int u = idx & 511, hd = idx >> 9;
  int h = hd >> 6, d = hd & 63;
  Wt[(long)hd * U_ + u] = f2bf(W[((long)h * U_ + u) * D_ + d] * scale);
}

__global__ void rowmax_kernel(const float* __restrict__ nef, float* __restrict__ mi) {
  __shared__ float red[256];
  int b = blockIdx.x, tid = threadIdx.x;
  red[tid] = nef[b * S_ + tid];
  __syncthreads();
  for (int s = 128; s > 0; s >>= 1) {
    if (tid < s) red[tid] = fmaxf(red[tid], red[tid + s]);
    __syncthreads();
  }
  if (tid == 0) mi[b] = fmaxf(red[0], 1.0f);
}

__global__ void posadd_kernel(float* __restrict__ x, unsigned short* __restrict__ xbf,
                              const float* __restrict__ nef,
                              const float* __restrict__ mi, const float* __restrict__ pe) {
  int bs = blockIdx.x;
  int b = bs >> 8;
  float nv = nef[bs];
  int pos;
  if (nv == -1.0f) pos = S_;
  else {
    pos = (int)(nv / mi[b] * (float)S_);
    if (pos < 0) pos = 0;
    if (pos > S_) pos = S_;
  }
  for (int u = threadIdx.x; u < U_; u += 256) {
    float v = x[(long)bs * U_ + u] + pe[pos * U_ + u];
    x[(long)bs * U_ + u]   = v;
    xbf[(long)bs * U_ + u] = f2bf(v);
  }
}

__global__ __launch_bounds__(256) void pool_kernel(const float* __restrict__ x,
    const float* __restrict__ nef, float* __restrict__ pooled) {
  int b = blockIdx.x, tid = threadIdx.x;
  float msum = 0.0f;
  for (int t = 0; t < S_; ++t) msum += (nef[b * S_ + t] != -1.0f) ? 1.0f : 0.0f;
  msum = fmaxf(msum, 1e-9f);
  for (int u = tid; u < U_; u += 256) {
    float s = 0.0f;
    for (int t = 0; t < S_; ++t) {
      float m = (nef[b * S_ + t] != -1.0f) ? 1.0f : 0.0f;
      s += x[((long)(b * S_ + t)) * U_ + u] * m;
    }
    pooled[b * U_ + u] = s / msum;
  }
}

// ---------------------------------------------------------------------------
extern "C" void kernel_launch(void* const* d_in, const int* in_sizes, int n_in,
                              void* d_out, int out_size, void* d_ws, size_t ws_size,
                              hipStream_t stream)
{
  (void)in_sizes; (void)n_in; (void)out_size; (void)ws_size;
  const float* frames     = (const float*)d_in[0];
  const float* nef        = (const float*)d_in[1];
  const float* lips_mean  = (const float*)d_in[2];
  const float* lips_std   = (const float*)d_in[3];
  const float* hand_mean  = (const float*)d_in[4];
  const float* hand_std   = (const float*)d_in[5];
  const float* pose_mean  = (const float*)d_in[6];
  const float* pose_std   = (const float*)d_in[7];
  const float* lips_empty = (const float*)d_in[8];
  const float* lips_W1    = (const float*)d_in[9];
  const float* lips_W2    = (const float*)d_in[10];
  const float* hand_empty = (const float*)d_in[11];
  const float* hand_W1    = (const float*)d_in[12];
  const float* hand_W2    = (const float*)d_in[13];
  const float* pose_empty = (const float*)d_in[14];
  const float* pose_W1    = (const float*)d_in[15];
  const float* pose_W2    = (const float*)d_in[16];
  const float* lm_w       = (const float*)d_in[17];
  const float* fc_W1      = (const float*)d_in[18];
  const float* fc_W2      = (const float*)d_in[19];
  const float* pos_emb    = (const float*)d_in[20];
  const float* Wq         = (const float*)d_in[21];
  const float* bq         = (const float*)d_in[22];
  const float* Wk         = (const float*)d_in[23];
  const float* bk         = (const float*)d_in[24];
  const float* Wv         = (const float*)d_in[25];
  const float* bv         = (const float*)d_in[26];
  const float* Wo         = (const float*)d_in[27];
  const float* bo         = (const float*)d_in[28];
  const float* Wm1        = (const float*)d_in[29];
  const float* bm1        = (const float*)d_in[30];
  const float* Wm2        = (const float*)d_in[31];
  const float* bm2        = (const float*)d_in[32];
  const float* clf_W      = (const float*)d_in[33];
  const float* clf_b      = (const float*)d_in[34];
  float* out = (float*)d_out;

  char* base = (char*)d_ws;
  size_t off = 0;
  auto allocf = [&](size_t n) { float* p = (float*)(base + off); off = (off + n*4 + 15) & ~(size_t)15; return p; };
  auto alloch = [&](size_t n) { unsigned short* p = (unsigned short*)(base + off); off = (off + n*2 + 15) & ~(size_t)15; return p; };

  float* fl     = allocf((size_t)BS_TOK * 80);
  float* fh     = allocf((size_t)BS_TOK * 42);
  float* fp     = allocf((size_t)BS_TOK * 10);
  float* miss   = allocf((size_t)BS_TOK * 3);
  float* xb     = allocf((size_t)BS_TOK * U_);
  float* pooled = allocf((size_t)B_ * U_);
  float* mi     = allocf(B_);
  unsigned short* t1b = alloch((size_t)BS_TOK * U_);   // also MLP hidden chunk
  unsigned short* xbb = alloch((size_t)BS_TOK * U_);
  unsigned short* qbb = alloch((size_t)BS_TOK * U_);
  unsigned short* kbb = alloch((size_t)BS_TOK * U_);
  unsigned short* vbb = alloch((size_t)BS_TOK * U_);
  unsigned short* zbb = alloch((size_t)BS_TOK * U_);
  unsigned short* w2lt = alloch((size_t)U_ * U_);
  unsigned short* w2ht = alloch((size_t)U_ * U_);
  unsigned short* w2pt = alloch((size_t)U_ * U_);
  unsigned short* fc1t = alloch((size_t)U_ * U_);
  unsigned short* fc2t = alloch((size_t)U_ * U_);
  unsigned short* wqt  = alloch((size_t)L_ * U_ * U_);
  unsigned short* wkt  = alloch((size_t)L_ * U_ * U_);
  unsigned short* wvt  = alloch((size_t)L_ * U_ * U_);
  unsigned short* wot  = alloch((size_t)L_ * U_ * U_);
  unsigned short* wm1t = alloch((size_t)L_ * U_ * MU_);
  unsigned short* wm2t = alloch((size_t)L_ * MU_ * U_);

  auto gemmb = [&](const unsigned short* A, int lda, const unsigned short* Bt, int ldb,
                   float* C, unsigned short* Cbf, int ldc, int M, int N, int K,
                   const float* bias, float bscale, const float* resid, int act,
                   int mode, int group, const float* emp) {
    dim3 g(N / 128, M / 128);
    gemm_bf16_kernel<<<g, 256, 0, stream>>>(A, lda, Bt, ldb, C, Cbf, ldc, M, N, K,
                                            bias, bscale, resid, act, mode, group,
                                            lm_w, emp, miss);
  };
  auto gemme = [&](const float* A, int lda, const float* Bw, int ldb,
                   float* C, unsigned short* Cbf, int ldc, int M, int N, int K,
                   const float* bias, int act) {
    dim3 g((N + 127) / 128, (M + 127) / 128);
    gemm_edge_kernel<<<g, 256, 0, stream>>>(A, lda, Bw, ldb, C, Cbf, ldc, M, N, K, bias, act);
  };
  auto wconv = [&](const float* W, unsigned short* Wt, int K, int N, float sc) {
    wtconv_kernel<<<(K * N + 255) / 256, 256, 0, stream>>>(W, Wt, K, N, sc);
  };

  // 0. weight pre-transpose/convert to bf16 [N][K]
  wconv(lips_W2, w2lt, U_, U_, 1.0f);
  wconv(hand_W2, w2ht, U_, U_, 1.0f);
  wconv(pose_W2, w2pt, U_, U_, 1.0f);
  wconv(fc_W1,   fc1t, U_, U_, 1.0f);
  wconv(fc_W2,   fc2t, U_, U_, 1.0f);
  const int qk_grid = (U_ * U_ + 255) / 256;
  for (int l = 0; l < L_; ++l) {
    // softmax scale 1/sqrt(D) folded into Wq (and bq via bias_scale below)
    wqkvconv_kernel<<<qk_grid, 256, 0, stream>>>(Wq + (long)l * U_ * U_, wqt + (long)l * U_ * U_, 0.125f);
    wqkvconv_kernel<<<qk_grid, 256, 0, stream>>>(Wk + (long)l * U_ * U_, wkt + (long)l * U_ * U_, 1.0f);
    wqkvconv_kernel<<<qk_grid, 256, 0, stream>>>(Wv + (long)l * U_ * U_, wvt + (long)l * U_ * U_, 1.0f);
    wconv(Wo  + (long)l * U_ * U_,  wot  + (long)l * U_ * U_,  U_,  U_,  1.0f);
    wconv(Wm1 + (long)l * U_ * MU_, wm1t + (long)l * U_ * MU_, U_,  MU_, 1.0f);
    wconv(Wm2 + (long)l * MU_ * U_, wm2t + (long)l * MU_ * U_, MU_, U_,  1.0f);
  }

  // 1. normalize landmarks + missing flags
  preprocess_kernel<<<(BS_TOK + 255) / 256, 256, 0, stream>>>(
      frames, lips_mean, lips_std, hand_mean, hand_std, pose_mean, pose_std,
      fl, fh, fp, miss);

  // 2. landmark embeds (+fused softmax(lm_w)/missing combine into xb)
  gemme(fl, 80, lips_W1, U_, nullptr, t1b, U_, BS_TOK, U_, 80, nullptr, 1);
  gemmb(t1b, U_, w2lt, U_, xb, xbb, U_, BS_TOK, U_, U_, nullptr, 1.0f, nullptr, 0, 2, 0, lips_empty);
  gemme(fh, 42, hand_W1, U_, nullptr, t1b, U_, BS_TOK, U_, 42, nullptr, 1);
  gemmb(t1b, U_, w2ht, U_, xb, xbb, U_, BS_TOK, U_, U_, nullptr, 1.0f, nullptr, 0, 2, 1, hand_empty);
  gemme(fp, 10, pose_W1, U_, nullptr, t1b, U_, BS_TOK, U_, 10, nullptr, 1);
  gemmb(t1b, U_, w2pt, U_, xb, xbb, U_, BS_TOK, U_, U_, nullptr, 1.0f, nullptr, 0, 2, 2, pose_empty);

  // 3. fc block
  gemmb(xbb, U_, fc1t, U_, nullptr, t1b, U_, BS_TOK, U_, U_, nullptr, 1.0f, nullptr, 1, 0, 0, nullptr);
  gemmb(t1b, U_, fc2t, U_, xb, xbb, U_, BS_TOK, U_, U_, nullptr, 1.0f, nullptr, 0, 0, 0, nullptr);

  // 4. positional embedding
  rowmax_kernel<<<B_, 256, 0, stream>>>(nef, mi);
  posadd_kernel<<<BS_TOK, 256, 0, stream>>>(xb, xbb, nef, mi, pos_emb);

  // 5. transformer layers
  for (int l = 0; l < L_; ++l) {
    gemmb(xbb, U_, wqt + (long)l * U_ * U_, U_, nullptr, qbb, U_, BS_TOK, U_, U_,
          bq + l * U_, 0.125f, nullptr, 0, 0, 0, nullptr);
    gemmb(xbb, U_, wkt + (long)l * U_ * U_, U_, nullptr, kbb, U_, BS_TOK, U_, U_,
          bk + l * U_, 1.0f, nullptr, 0, 0, 0, nullptr);
    gemmb(xbb, U_, wvt + (long)l * U_ * U_, U_, nullptr, vbb, U_, BS_TOK, U_, U_,
          bv + l * U_, 1.0f, nullptr, 0, 0, 0, nullptr);
    attn_kernel<<<B_ * H_, 256, 0, stream>>>(qbb, kbb, vbb, nef, zbb);
    gemmb(zbb, U_, wot + (long)l * U_ * U_, U_, xb, xbb, U_, BS_TOK, U_, U_,
          bo + l * U_, 1.0f, xb, 0, 0, 0, nullptr);
    for (int ch = 0; ch < 4; ++ch) {       // MLP chunked: hidden lives in t1b
      float*          xc  = xb  + (long)ch * 8192 * U_;
      unsigned short* xcb = xbb + (long)ch * 8192 * U_;
      gemmb(xcb, U_, wm1t + (long)l * U_ * MU_, U_, nullptr, t1b, MU_, 8192, MU_, U_,
            bm1 + l * MU_, 1.0f, nullptr, 1, 0, 0, nullptr);
      gemmb(t1b, MU_, wm2t + (long)l * MU_ * U_, MU_, xc, xcb, U_, 8192, U_, MU_,
            bm2 + l * U_, 1.0f, xc, 0, 0, 0, nullptr);
    }
  }

  // 6. masked mean pool + classifier
  pool_kernel<<<B_, 256, 0, stream>>>(xb, nef, pooled);
  gemme(pooled, U_, clf_W, NC_, out, nullptr, NC_, B_, NC_, U_, clf_b, 0);
}
